// LSTM_68126771249674
// MI455X (gfx1250) — matrix operand order repaired and checked by reading, not compiled, not run on hardware
//
#include <hip/hip_runtime.h>
#include <hip/hip_bf16.h>
#include <math.h>

// ---------------------------------------------------------------------------
// LSTM (T=128,B=32,D=512,H=1024,V=32000) on gfx1250.
// bf16 WMMA (v_wmma_f32_16x16x32_bf16) for all GEMMs, f32 accumulate.
// Block = one 16-row M slab staged in LDS; wave = 16x64 output (4 N tiles);
// K loop unrolled x2 with ping-pong register buffers; K is a template
// constant so B-tile addresses fold into instruction immediate offsets.
// ---------------------------------------------------------------------------

typedef __attribute__((ext_vector_type(16))) __bf16 v16bf;
typedef __attribute__((ext_vector_type(8)))  __bf16 v8bf;
typedef __attribute__((ext_vector_type(8)))  float  v8f;

#define TT 128
#define BB 32
#define DD 512
#define HH 1024
#define VV 32000

static __device__ __forceinline__ __bf16 f32_to_bf16(float f) {
  unsigned u = __builtin_bit_cast(unsigned, f);
  u += 0x7FFFu + ((u >> 16) & 1u);           // round-to-nearest-even
  unsigned short s = (unsigned short)(u >> 16);
  return __builtin_bit_cast(__bf16, s);
}

// ---- fragment lane-base helpers (computed once, indexed by k offset) ------

// A fragment (16x32 MxK): lane L<16 -> row L, K=0..15 contiguous;
// lane L>=16 -> row L-16, K=16..31.  Base points at this lane's k=0.
static __device__ __forceinline__ const __bf16* a_lane_base(const __bf16* Alds, int ldl) {
  int lane = threadIdx.x & 31;
  return Alds + (lane & 15) * ldl + ((lane >> 4) << 4);
}
static __device__ __forceinline__ v16bf load_a_at(const __bf16* lb, int k0) {
  v8bf lo = *(const v8bf*)(lb + k0);         // ds_load_b128
  v8bf hi = *(const v8bf*)(lb + k0 + 8);
  v16bf a;
#pragma unroll
  for (int i = 0; i < 8; ++i) { a[i] = lo[i]; a[i + 8] = hi[i]; }
  return a;
}

// B fragment (32x16 KxN), B[k][n] = W[n0+n][k] (weights row-major NxK):
// lane L<16 holds col N=L, K=k0+0..15 contiguous; lane L>=16: K=k0+16..31.
static __device__ __forceinline__ const __bf16* b_lane_base(const __bf16* __restrict__ W,
                                                            int ld, int n0) {
  int lane = threadIdx.x & 31;
  return W + (size_t)(n0 + (lane & 15)) * ld + ((lane >> 4) << 4);
}
static __device__ __forceinline__ v16bf load_b_at(const __bf16* lb, int off) {
  return *(const v16bf*)(lb + off);          // 2x global_load_b128
}

// Cooperative global->LDS stage of one 16xK bf16 slab (row stride ldl = K+8
// elements: 2064B = 516 dwords -> rows rotate 4 banks, conflict-free b128).
template <int K>
static __device__ __forceinline__ void stage_a_slab(const __bf16* __restrict__ g,
                                                    __bf16* Alds, int ldl, int m0) {
  constexpr int kpr = K >> 3;                // 16B chunks per row
  constexpr int chunks = 16 * kpr;
  for (int idx = threadIdx.x; idx < chunks; idx += blockDim.x) {
    int row = idx / kpr;
    int kc  = (idx % kpr) << 3;
    *(v8bf*)(Alds + row * ldl + kc) = *(const v8bf*)(g + (size_t)(m0 + row) * K + kc);
  }
}

// --------------------------- elementwise helpers ---------------------------

__global__ void cvt_f32_to_bf16(const float* __restrict__ in, __bf16* __restrict__ out,
                                long long n) {
  long long i = (long long)blockIdx.x * blockDim.x + threadIdx.x;
  if (i < n) out[i] = f32_to_bf16(in[i]);
}

__global__ void copy_f32(const float* __restrict__ in, float* __restrict__ out, int n) {
  int i = blockIdx.x * blockDim.x + threadIdx.x;
  if (i < n) out[i] = in[i];
}

__global__ void tail_copy(const float* __restrict__ hcur, const float* __restrict__ cbuf,
                          float* __restrict__ dst, int n) {
  int i = blockIdx.x * blockDim.x + threadIdx.x;
  if (i < n) { dst[i] = hcur[i]; dst[n + i] = cbuf[i]; }
}

// ------------------------------- GEMM (WMMA) -------------------------------
// out[m][n] = sum_k A[m][k]*W[n][k] + bias0[n] + bias1[n]
// grid.y = M tile; block stages A slab in LDS; each wave: 4 N tiles.
// K: compile-time (512 or 1024), multiple of 64.

template <int K>
__global__ void gemm_bf16_wmma(const __bf16* __restrict__ A, const __bf16* __restrict__ W,
                               const float* __restrict__ bias0, const float* __restrict__ bias1,
                               float* __restrict__ outF, __bf16* __restrict__ outB,
                               int N, int nslabs) {
  constexpr int ldl = K + 8;
  __shared__ __bf16 Alds[16 * ldl];
  const int m0 = blockIdx.y * 16;

  stage_a_slab<K>(A, Alds, ldl, m0);
  __syncthreads();

  int slab = blockIdx.x * (blockDim.x >> 5) + (threadIdx.x >> 5);
  if (slab >= nslabs) return;                // wave-uniform, after barrier
  int n0 = slab * 64;                        // 4 consecutive 16-col tiles

  const __bf16* ab = a_lane_base(Alds, ldl);
  const __bf16* bb = b_lane_base(W, K, n0);  // tiles at constant offsets q*16*K

  v8f acc[4] = {{}, {}, {}, {}};
  v16bf a0, a1, b0[4], b1[4];
  a0 = load_a_at(ab, 0);
#pragma unroll
  for (int q = 0; q < 4; ++q) b0[q] = load_b_at(bb, q * 16 * K);

#pragma unroll 1
  for (int k0 = 0; k0 < K; k0 += 64) {       // 2 k-steps per iteration
    a1 = load_a_at(ab, k0 + 32);             // stage odd step
#pragma unroll
    for (int q = 0; q < 4; ++q) b1[q] = load_b_at(bb, q * 16 * K + k0 + 32);
#pragma unroll
    for (int q = 0; q < 4; ++q)
      acc[q] = __builtin_amdgcn_wmma_f32_16x16x32_bf16(false, a0, false, b0[q],
                                                       (short)0, acc[q], false, false);
    if (k0 + 64 < K) {                       // stage next even step
      a0 = load_a_at(ab, k0 + 64);
#pragma unroll
      for (int q = 0; q < 4; ++q) b0[q] = load_b_at(bb, q * 16 * K + k0 + 64);
    }
#pragma unroll
    for (int q = 0; q < 4; ++q)
      acc[q] = __builtin_amdgcn_wmma_f32_16x16x32_bf16(false, a1, false, b1[q],
                                                       (short)0, acc[q], false, false);
  }

  int lane = threadIdx.x & 31;
  int mh   = m0 + ((lane >> 4) << 3);
#pragma unroll
  for (int q = 0; q < 4; ++q) {
    int n = n0 + q * 16 + (lane & 15);
    float bias = (bias0 ? bias0[n] : 0.f) + (bias1 ? bias1[n] : 0.f);
#pragma unroll
    for (int r = 0; r < 8; ++r) {
      float v = acc[q][r] + bias;
      size_t o = (size_t)(mh + r) * N + n;
      if (outF) outF[o] = v;
      if (outB) outB[o] = f32_to_bf16(v);
    }
  }
}

// ------------------------- fused LSTM recurrence step ----------------------
// gates[b][q*H+n] = xp_t[b][q*H+n] + sum_k h[b][k]*W_hh[q*H+n][k]
// grid = (8, 2): block stages its 16-row h slab in LDS; each wave owns one
// 16-col slice of all four gates, applies i/f/g/o update in-register.

__global__ void lstm_step(const float* __restrict__ xp_t, const __bf16* __restrict__ Whh,
                          const __bf16* __restrict__ h_in, __bf16* __restrict__ h_out,
                          float* __restrict__ cbuf, float* __restrict__ hcur,
                          __bf16* __restrict__ hs_t) {
  constexpr int ldl = HH + 8;
  __shared__ __bf16 Alds[16 * ldl];
  const int m0 = blockIdx.y * 16;

  stage_a_slab<HH>(h_in, Alds, ldl, m0);
  __syncthreads();

  int nt = blockIdx.x * (blockDim.x >> 5) + (threadIdx.x >> 5);  // 0..63
  int n0 = nt * 16;

  const __bf16* ab = a_lane_base(Alds, ldl);
  const __bf16* bb = b_lane_base(Whh, HH, n0);   // gate q at offset q*HH*HH

  v8f acc[4] = {{}, {}, {}, {}};
  v16bf a0, a1, b0[4], b1[4];
  a0 = load_a_at(ab, 0);
#pragma unroll
  for (int q = 0; q < 4; ++q) b0[q] = load_b_at(bb, q * HH * HH);

#pragma unroll 1
  for (int k0 = 0; k0 < HH; k0 += 64) {
    a1 = load_a_at(ab, k0 + 32);
#pragma unroll
    for (int q = 0; q < 4; ++q) b1[q] = load_b_at(bb, q * HH * HH + k0 + 32);
#pragma unroll
    for (int q = 0; q < 4; ++q)
      acc[q] = __builtin_amdgcn_wmma_f32_16x16x32_bf16(false, a0, false, b0[q],
                                                       (short)0, acc[q], false, false);
    if (k0 + 64 < HH) {
      a0 = load_a_at(ab, k0 + 64);
#pragma unroll
      for (int q = 0; q < 4; ++q) b0[q] = load_b_at(bb, q * HH * HH + k0 + 64);
    }
#pragma unroll
    for (int q = 0; q < 4; ++q)
      acc[q] = __builtin_amdgcn_wmma_f32_16x16x32_bf16(false, a1, false, b1[q],
                                                       (short)0, acc[q], false, false);
  }

  int lane = threadIdx.x & 31;
  int n  = n0 + (lane & 15);
  int mh = m0 + ((lane >> 4) << 3);
#pragma unroll
  for (int r = 0; r < 8; ++r) {
    int m = mh + r;
    const float* xr = xp_t + (size_t)m * (4 * HH);
    float gi = acc[0][r] + xr[0 * HH + n];
    float gf = acc[1][r] + xr[1 * HH + n];
    float gg = acc[2][r] + xr[2 * HH + n];
    float go = acc[3][r] + xr[3 * HH + n];
    float si = 1.f / (1.f + expf(-gi));
    float sf = 1.f / (1.f + expf(-gf));
    float so = 1.f / (1.f + expf(-go));
    int ci = m * HH + n;                     // each element owned by one wave
    float c_new = sf * cbuf[ci] + si * tanhf(gg);
    float h_new = so * tanhf(c_new);
    cbuf[ci] = c_new;
    hcur[ci] = h_new;
    __bf16 hb = f32_to_bf16(h_new);
    h_out[ci] = hb;                          // next step's A operand
    hs_t[ci]  = hb;                          // for fc_in GEMM
  }
}

// --------------------------------- launch ----------------------------------

static inline size_t align_up(size_t x) { return (x + 255) & ~(size_t)255; }

extern "C" void kernel_launch(void* const* d_in, const int* in_sizes, int n_in,
                              void* d_out, int out_size, void* d_ws, size_t ws_size,
                              hipStream_t stream) {
  (void)in_sizes; (void)n_in; (void)out_size; (void)ws_size;

  const float* feature  = (const float*)d_in[0];   // [T,B,D]
  const float* h0       = (const float*)d_in[1];   // [1,B,H]
  const float* c0       = (const float*)d_in[2];   // [1,B,H]
  const float* W_ih     = (const float*)d_in[3];   // [4H,D]
  const float* W_hh     = (const float*)d_in[4];   // [4H,H]
  const float* b_ih     = (const float*)d_in[5];   // [4H]
  const float* b_hh     = (const float*)d_in[6];   // [4H]
  const float* fc_in_w  = (const float*)d_in[7];   // [H,H]
  const float* fc_in_b  = (const float*)d_in[8];   // [H]
  const float* fc_out_w = (const float*)d_in[9];   // [V,H]
  const float* fc_out_b = (const float*)d_in[10];  // [V]
  float* out = (float*)d_out;

  const long long featN  = (long long)TT * BB * DD;
  const long long wihN   = (long long)4 * HH * DD;
  const long long whhN   = (long long)4 * HH * HH;
  const long long fcinN  = (long long)HH * HH;
  const long long fcoutN = (long long)VV * HH;
  const long long xpN    = (long long)TT * BB * 4 * HH;
  const int       hN     = BB * HH;
  const long long hsN    = (long long)TT * BB * HH;

  char* p = (char*)d_ws; size_t off = 0;
  auto carve = [&](size_t bytes) { char* r = p + off; off = align_up(off + bytes); return r; };
  __bf16* feat_bf  = (__bf16*)carve(featN  * 2);
  __bf16* wih_bf   = (__bf16*)carve(wihN   * 2);
  __bf16* whh_bf   = (__bf16*)carve(whhN   * 2);
  __bf16* fcin_bf  = (__bf16*)carve(fcinN  * 2);
  __bf16* fcout_bf = (__bf16*)carve(fcoutN * 2);
  float*  xp       = (float*) carve(xpN    * 4);
  __bf16* hbuf0    = (__bf16*)carve((size_t)hN * 2);
  __bf16* hbuf1    = (__bf16*)carve((size_t)hN * 2);
  float*  cbuf     = (float*) carve((size_t)hN * 4);
  float*  hcur     = (float*) carve((size_t)hN * 4);
  __bf16* hs_bf    = (__bf16*)carve(hsN * 2);
  __bf16* tmp_bf   = (__bf16*)carve(hsN * 2);

  const int THR = 256;
  auto gsz = [&](long long n) { return (unsigned)((n + THR - 1) / THR); };

  // 1) precision conversion of matmul operands (f32 -> bf16)
  cvt_f32_to_bf16<<<gsz(featN),  THR, 0, stream>>>(feature,  feat_bf,  featN);
  cvt_f32_to_bf16<<<gsz(wihN),   THR, 0, stream>>>(W_ih,     wih_bf,   wihN);
  cvt_f32_to_bf16<<<gsz(whhN),   THR, 0, stream>>>(W_hh,     whh_bf,   whhN);
  cvt_f32_to_bf16<<<gsz(fcinN),  THR, 0, stream>>>(fc_in_w,  fcin_bf,  fcinN);
  cvt_f32_to_bf16<<<gsz(fcoutN), THR, 0, stream>>>(fc_out_w, fcout_bf, fcoutN);
  cvt_f32_to_bf16<<<gsz(hN),     THR, 0, stream>>>(h0,       hbuf0,    hN);
  copy_f32<<<gsz(hN), THR, 0, stream>>>(c0, cbuf, hN);

  // 2) xp = feature @ W_ih^T + b_ih + b_hh  (M=4096, N=4096, K=512)
  {
    int nslabs = (4 * HH / 16) / 4;                      // 64
    dim3 grid((nslabs + 7) / 8, TT * BB / 16);           // (8, 256)
    gemm_bf16_wmma<DD><<<grid, THR, 0, stream>>>(feat_bf, wih_bf, b_ih, b_hh,
                                                 xp, (__bf16*)nullptr, 4 * HH, nslabs);
  }

  // 3) recurrence: 128 dependent launches, fused GEMM + gates
  for (int t = 0; t < TT; ++t) {
    const __bf16* hin  = (t & 1) ? hbuf1 : hbuf0;
    __bf16*       hout = (t & 1) ? hbuf0 : hbuf1;
    lstm_step<<<dim3(8, 2), THR, 0, stream>>>(xp + (size_t)t * BB * 4 * HH, whh_bf,
                                              hin, hout, cbuf, hcur,
                                              hs_bf + (size_t)t * BB * HH);
  }

  // 4) tmp = hs @ fc_in_w^T + fc_in_b  (M=4096, N=1024, K=1024), bf16 out
  {
    int nslabs = (HH / 16) / 4;                          // 16
    dim3 grid((nslabs + 7) / 8, TT * BB / 16);           // (2, 256)
    gemm_bf16_wmma<HH><<<grid, THR, 0, stream>>>(hs_bf, fcin_bf, fc_in_b, (const float*)nullptr,
                                                 (float*)nullptr, tmp_bf, HH, nslabs);
  }

  // 5) out = tmp @ fc_out_w^T + fc_out_b  (M=4096, N=32000, K=1024), f32 out
  {
    int nslabs = (VV / 16) / 4;                          // 500
    dim3 grid((nslabs + 7) / 8, TT * BB / 16);           // (63, 256)
    gemm_bf16_wmma<HH><<<grid, THR, 0, stream>>>(tmp_bf, fcout_bf, fc_out_b, (const float*)nullptr,
                                                 out, (__bf16*)nullptr, VV, nslabs);
  }

  // 6) tuple tail: h_T then c_T
  tail_copy<<<gsz(hN), THR, 0, stream>>>(hcur, cbuf, out + (size_t)TT * BB * VV, hN);
}
